// MIL_Cluster_FC_surv_43782896615746
// MI455X (gfx1250) — matrix operand order repaired and checked by reading, not compiled, not run on hardware
//
#include <hip/hip_runtime.h>
#include <hip/hip_bf16.h>

// ---------------------------------------------------------------------------
// MIL_Cluster_FC_surv for MI455X (gfx1250, wave32, WMMA f16 16x16x32,
// async global->LDS staging for activations, LDS-staged f16 weights)
// ---------------------------------------------------------------------------

#define NPATCH 8192
#define KCLUS  10
#define DIN    1024
#define DH     512
#define DA     256
#define NCLS   4
#define ROWS_PER_BLOCK 32
#define MAX_TILES 256          // ceil(NPATCH / ROWS_PER_BLOCK)
#define MAX_SLOTS 272          // sum_k ceil(cnt_k/32) <= 256 + 9, padded

// LDS layout strides (chosen for 16B alignment + conflict-free fragment reads)
#define XS_STRIDE_F32 36       // 144 B per row  (32 f32 + 4 pad)
#define WB_STRIDE_H   40       // 80 B per column (32 f16 + 8 pad)
#define HS_STRIDE_H   520      // 1040 B per row (512 f16 + 8 pad)

typedef __attribute__((ext_vector_type(16))) _Float16 v16h;
typedef __attribute__((ext_vector_type(8)))  _Float16 v8h;
typedef __attribute__((ext_vector_type(2)))  _Float16 v2h;
typedef __attribute__((ext_vector_type(8)))  float    v8f;

// ---------------------------------------------------------------------------
// Kernel 1: deterministic bucket sort of patch indices by cluster (1 wave32).
// ---------------------------------------------------------------------------
__global__ void bucket_kernel(const int* __restrict__ cid,
                              int* __restrict__ order,
                              int* __restrict__ cnt,
                              int* __restrict__ offs,
                              int* __restrict__ tileoff) {
    __shared__ int cur[16];
    const int lane = threadIdx.x;   // 0..31

    int total = 0;
    for (int base = 0; base < NPATCH; base += 32) {
        int myc = cid[base + lane];
        for (int cc = 0; cc < KCLUS; ++cc) {
            unsigned long long m = __ballot(myc == cc);
            if (lane == cc) total += __popcll(m);
        }
    }
    if (lane < KCLUS) cnt[lane] = total;
    __syncthreads();

    if (lane == 0) {
        int o = 0, to = 0;
        for (int cc = 0; cc < KCLUS; ++cc) {
            offs[cc] = o;
            tileoff[cc] = to;
            int n = cnt[cc];
            cur[cc] = o;
            o  += n;
            to += (n + ROWS_PER_BLOCK - 1) / ROWS_PER_BLOCK;
        }
        offs[KCLUS] = o;
        tileoff[KCLUS] = to;
    }
    __syncthreads();

    for (int base = 0; base < NPATCH; base += 32) {
        int myc = cid[base + lane];
        int pos = 0;
        for (int cc = 0; cc < KCLUS; ++cc) {
            unsigned long long m64 = __ballot(myc == cc);
            unsigned mm = (unsigned)m64;
            if (myc == cc) {
                unsigned ltmask = (lane == 0) ? 0u : (0xFFFFFFFFu >> (32 - lane));
                pos = cur[cc] + __popc(mm & ltmask);
            }
            __syncthreads();
            if (lane == cc) cur[cc] += __popc(mm);
            __syncthreads();
        }
        order[pos] = base + lane;
    }
}

// ---------------------------------------------------------------------------
// Kernel 2: fused per-cluster 2-layer MLP over 32 gathered rows, WMMA f16.
// grid = (MAX_TILES, KCLUS), block = 256 (8 waves).
// ---------------------------------------------------------------------------
__global__ void __launch_bounds__(256)
phi_kernel(const float* __restrict__ x,
           const int*   __restrict__ order,
           const int*   __restrict__ cnt,
           const int*   __restrict__ offs,
           const int*   __restrict__ tileoff,
           const float* __restrict__ W1,
           const float* __restrict__ b1,
           const float* __restrict__ W2,
           const float* __restrict__ b2,
           float* __restrict__ partials) {
    const int c = blockIdx.y;
    const int t = blockIdx.x;
    const int n_k = cnt[c];
    const int ntile = (n_k + ROWS_PER_BLOCK - 1) / ROWS_PER_BLOCK;
    if (t >= ntile) return;                 // uniform exit: EXEC stays full

    __shared__ __align__(16) _Float16 Hs[ROWS_PER_BLOCK * HS_STRIDE_H]; // 33280 B
    __shared__ __align__(16) _Float16 Wb[DH * WB_STRIDE_H];             // 40960 B  [n][k] f16
    __shared__ __align__(16) float    Xs[ROWS_PER_BLOCK * XS_STRIDE_F32]; // 4608 B [row][k] f32
    __shared__ float Cs[2][DH];                                          // 4096 B

    const int slot = tileoff[c] + t;
    const int p0   = offs[c] + t * ROWS_PER_BLOCK;

    const int tid  = threadIdx.x;
    const int lane = tid & 31;
    const int wave = tid >> 5;
    const int rt   = wave & 1;              // row tile 0/1
    const int cg   = wave >> 1;             // col group 0..3 (128 cols each)
    const int mrow = lane & 15;
    const bool hi  = lane >= 16;

    // ---- staging roles ----
    // X async stage: thread -> (row 0..31, 16B chunk 0..7 of the 128B k-slab)
    const int  srow   = tid >> 3;
    const int  schk   = tid & 7;
    const bool svalid = (t * ROWS_PER_BLOCK + srow) < n_k;
    const int  sgsrc  = svalid ? order[p0 + srow] : 0;
    const unsigned long long sx_chunk =
        (unsigned long long)(const void*)(x + (size_t)sgsrc * DIN + schk * 4);
    const unsigned xs_lds =
        (unsigned)(uintptr_t)&Xs[srow * XS_STRIDE_F32] + (unsigned)(schk * 16);
    // W stage: thread -> 4 consecutive n at (tid&127)*4, k-half (tid>>7)*16
    const int wn0 = (tid & 127) * 4;
    const int wkh = (tid >> 7) * 16;

    const float* W1k = W1 + (size_t)c * DIN * DH;
    const float* W2k = W2 + (size_t)c * DH * DH;
    const float* b1k = b1 + c * DH;
    const float* b2k = b2 + c * DH;

    // stage one 32-deep k-slab of a [*,DH] f32 weight matrix into Wb as f16 [n][k]
    auto stage_w = [&](const float* __restrict__ Wk, int k0) {
        #pragma unroll
        for (int i = 0; i < 16; i += 2) {
            const int k = k0 + wkh + i;
            const float4 va = *(const float4*)(Wk + (size_t)k * DH + wn0);
            const float4 vb = *(const float4*)(Wk + (size_t)(k + 1) * DH + wn0);
            const float av[4] = {va.x, va.y, va.z, va.w};
            const float bv[4] = {vb.x, vb.y, vb.z, vb.w};
            #pragma unroll
            for (int n = 0; n < 4; ++n) {
                v2h p;
                p[0] = (_Float16)av[n];
                p[1] = (_Float16)bv[n];
                *(v2h*)&Wb[(wn0 + n) * WB_STRIDE_H + wkh + i] = p;
            }
        }
    };

    // fetch a B fragment (32x16) for column tile base n from Wb
    auto load_b = [&](int n) -> v16h {
        const _Float16* wp = &Wb[n * WB_STRIDE_H + (hi ? 8 : 0)];
        v8h lo = *(const v8h*)wp;
        v8h hh = *(const v8h*)(wp + 16);
        return __builtin_shufflevector(lo, hh, 0, 1, 2, 3, 4, 5, 6, 7,
                                               8, 9, 10, 11, 12, 13, 14, 15);
    };

    v8f acc[8] = {};
    const int Boff = hi ? 8 : 0;   // A-fragment k-chunk base within 32-slab

    // ---------------- layer 1: H = relu(Xg @ W1k + b1k) ----------------
    for (int kt = 0; kt < DIN / 32; ++kt) {
        const int k0 = kt * 32;
        __syncthreads();   // previous slab consumed; Xs/Wb free

        // async-copy X slab (32 rows x 32 k, f32) straight into LDS
        asm volatile("global_load_async_to_lds_b128 %0, %1, off"
                     :: "v"(xs_lds),
                        "v"(sx_chunk + (unsigned long long)k0 * 4)
                     : "memory");
        // stage W1 slab (f32 -> f16, transposed to [n][k])
        stage_w(W1k, k0);

        asm volatile("s_wait_asynccnt 0x0" ::: "memory");
        __syncthreads();   // staged data visible

        // A fragment: 4x ds_load_b128 (f32) + packed cvt
        const float* xrow = &Xs[(rt * 16 + mrow) * XS_STRIDE_F32];
        const float4 r0 = *(const float4*)(xrow + Boff);
        const float4 r1 = *(const float4*)(xrow + Boff + 4);
        const float4 r2 = *(const float4*)(xrow + Boff + 16);
        const float4 r3 = *(const float4*)(xrow + Boff + 20);
        v16h a;
        a[0]  = (_Float16)r0.x; a[1]  = (_Float16)r0.y;
        a[2]  = (_Float16)r0.z; a[3]  = (_Float16)r0.w;
        a[4]  = (_Float16)r1.x; a[5]  = (_Float16)r1.y;
        a[6]  = (_Float16)r1.z; a[7]  = (_Float16)r1.w;
        a[8]  = (_Float16)r2.x; a[9]  = (_Float16)r2.y;
        a[10] = (_Float16)r2.z; a[11] = (_Float16)r2.w;
        a[12] = (_Float16)r3.x; a[13] = (_Float16)r3.y;
        a[14] = (_Float16)r3.z; a[15] = (_Float16)r3.w;

        #pragma unroll
        for (int ct = 0; ct < 8; ++ct) {
            const int n = cg * 128 + ct * 16;
            const v16h b = load_b(n + mrow);
            acc[ct] = __builtin_amdgcn_wmma_f32_16x16x32_f16(
                false, a, false, b, (short)0, acc[ct], false, false);
        }
    }

    // epilogue 1: bias + relu -> Hs (f16)
    #pragma unroll
    for (int ct = 0; ct < 8; ++ct) {
        const int n = cg * 128 + ct * 16 + mrow;
        const float bias = b1k[n];
        #pragma unroll
        for (int r = 0; r < 8; ++r) {
            const int m = rt * 16 + r + (hi ? 8 : 0);
            float v = acc[ct][r] + bias;
            v = v > 0.f ? v : 0.f;
            Hs[m * HS_STRIDE_H + n] = (_Float16)v;
        }
    }

    // ---------------- layer 2: Y = relu(H @ W2k + b2k) ----------------
    #pragma unroll
    for (int i = 0; i < 8; ++i) acc[i] = (v8f){};

    for (int kt = 0; kt < DH / 32; ++kt) {
        const int k0 = kt * 32;
        __syncthreads();               // Wb free (and Hs complete on first iter)
        stage_w(W2k, k0);
        __syncthreads();               // staged data visible

        // A fragment straight from Hs (f16): 2x v8h LDS loads
        const _Float16* hrow = &Hs[(rt * 16 + mrow) * HS_STRIDE_H + k0];
        v8h alo = *(const v8h*)(hrow + Boff);
        v8h ahh = *(const v8h*)(hrow + Boff + 16);
        v16h a = __builtin_shufflevector(alo, ahh, 0, 1, 2, 3, 4, 5, 6, 7,
                                                   8, 9, 10, 11, 12, 13, 14, 15);
        #pragma unroll
        for (int ct = 0; ct < 8; ++ct) {
            const int n = cg * 128 + ct * 16;
            const v16h b = load_b(n + mrow);
            acc[ct] = __builtin_amdgcn_wmma_f32_16x16x32_f16(
                false, a, false, b, (short)0, acc[ct], false, false);
        }
    }

    // epilogue 2: bias + relu, masked column sum over 32 rows
    #pragma unroll
    for (int ct = 0; ct < 8; ++ct) {
        const int n = cg * 128 + ct * 16 + mrow;
        const float bias = b2k[n];
        float s = 0.f;
        #pragma unroll
        for (int r = 0; r < 8; ++r) {
            const int m = rt * 16 + r + (hi ? 8 : 0);
            const bool rv = (t * ROWS_PER_BLOCK + m) < n_k;
            float y = acc[ct][r] + bias;
            y = y > 0.f ? y : 0.f;
            s += rv ? y : 0.f;
        }
        s += __shfl_xor(s, 16, 32);     // combine lane L with L^16 (same column)
        if (!hi) Cs[rt][n] = s;
    }
    __syncthreads();

    float* part = partials + (size_t)slot * DH;
    for (int j = tid; j < DH; j += 256)
        part[j] = Cs[0][j] + Cs[1][j];
}

// ---------------------------------------------------------------------------
// Kernel 3: deterministic reduction of tile partials -> pooled[K][512]
// ---------------------------------------------------------------------------
__global__ void reduce_kernel(const float* __restrict__ partials,
                              const int* __restrict__ cnt,
                              const int* __restrict__ tileoff,
                              float* __restrict__ pooled) {
    const int c = blockIdx.x;
    const int j = threadIdx.x;
    const int n = cnt[c];
    const int nt = (n + ROWS_PER_BLOCK - 1) / ROWS_PER_BLOCK;
    const int t0 = tileoff[c];
    float s = 0.f;
    for (int t = 0; t < nt; ++t) s += partials[(size_t)(t0 + t) * DH + j];
    pooled[c * DH + j] = (n > 0) ? (s / (float)n) : 0.f;
}

// ---------------------------------------------------------------------------
// Kernel 4: attention tail (tiny, fp32 VALU). 1 block, 512 threads.
// ---------------------------------------------------------------------------
__global__ void tail_kernel(const float* __restrict__ pooled,
                            const float* __restrict__ fcW, const float* __restrict__ fcb,
                            const float* __restrict__ Va,  const float* __restrict__ ba,
                            const float* __restrict__ Vb,  const float* __restrict__ bb,
                            const float* __restrict__ Vc,  const float* __restrict__ bc,
                            const float* __restrict__ rhoW, const float* __restrict__ rhob,
                            const float* __restrict__ clsW, const float* __restrict__ clsb,
                            float* __restrict__ out) {
    __shared__ float ps[KCLUS][DH];
    __shared__ float hs[KCLUS][DH];
    __shared__ float red[512];
    __shared__ float Ak[KCLUS];
    __shared__ float hp[DH];
    __shared__ float hr[DA];
    __shared__ float logits[NCLS];
    const int tid = threadIdx.x;

    for (int i = tid; i < KCLUS * DH; i += 512) ps[i / DH][i % DH] = pooled[i];
    __syncthreads();

    for (int kk = 0; kk < KCLUS; ++kk) {
        const int j = tid;
        float s = fcb[j];
        for (int i = 0; i < DH; ++i) s += ps[kk][i] * fcW[i * DH + j];
        hs[kk][j] = s > 0.f ? s : 0.f;
    }
    __syncthreads();

    for (int kk = 0; kk < KCLUS; ++kk) {
        float contrib = 0.f;
        if (tid < DA) {
            const int m = tid;
            float sa = ba[m], sb = bb[m];
            for (int i = 0; i < DH; ++i) {
                const float h = hs[kk][i];
                sa += h * Va[i * DA + m];
                sb += h * Vb[i * DA + m];
            }
            const float tn = tanhf(sa);
            const float sg = 1.f / (1.f + expf(-sb));
            contrib = tn * sg * Vc[m];
        }
        red[tid] = contrib;
        __syncthreads();
        for (int sft = 256; sft > 0; sft >>= 1) {
            if (tid < sft) red[tid] += red[tid + sft];
            __syncthreads();
        }
        if (tid == 0) Ak[kk] = red[0] + bc[0];
        __syncthreads();
    }

    if (tid == 0) {
        float mx = Ak[0];
        for (int i = 1; i < KCLUS; ++i) mx = Ak[i] > mx ? Ak[i] : mx;
        float ss = 0.f;
        for (int i = 0; i < KCLUS; ++i) { Ak[i] = expf(Ak[i] - mx); ss += Ak[i]; }
        for (int i = 0; i < KCLUS; ++i) Ak[i] /= ss;
    }
    __syncthreads();

    {
        const int j = tid;
        float s = 0.f;
        for (int kk = 0; kk < KCLUS; ++kk) s += Ak[kk] * hs[kk][j];
        hp[j] = s;
    }
    __syncthreads();

    if (tid < DA) {
        const int m = tid;
        float s = rhob[m];
        for (int i = 0; i < DH; ++i) s += hp[i] * rhoW[i * DA + m];
        hr[m] = s > 0.f ? s : 0.f;
    }
    __syncthreads();

    if (tid < NCLS) {
        float s = clsb[tid];
        for (int i = 0; i < DA; ++i) s += hr[i] * clsW[i * NCLS + tid];
        logits[tid] = s;
    }
    __syncthreads();

    if (tid == 0) {
        int best = 0; float bv = logits[0];
        for (int i = 1; i < NCLS; ++i) if (logits[i] > bv) { bv = logits[i]; best = i; }
        float cum = 1.f;
        for (int i = 0; i < NCLS; ++i) {
            const float hz = 1.f / (1.f + expf(-logits[i]));
            out[i] = hz;
            cum *= (1.f - hz);
            out[NCLS + i] = cum;
        }
        out[2 * NCLS] = (float)best;
    }
}

// ---------------------------------------------------------------------------
// Host launcher
// ---------------------------------------------------------------------------
extern "C" void kernel_launch(void* const* d_in, const int* in_sizes, int n_in,
                              void* d_out, int out_size, void* d_ws, size_t ws_size,
                              hipStream_t stream) {
    const float* x    = (const float*)d_in[0];
    const int*   cid  = (const int*)  d_in[1];
    const float* W1   = (const float*)d_in[2];
    const float* b1   = (const float*)d_in[3];
    const float* W2   = (const float*)d_in[4];
    const float* b2   = (const float*)d_in[5];
    const float* fcW  = (const float*)d_in[6];
    const float* fcb  = (const float*)d_in[7];
    const float* Va   = (const float*)d_in[8];
    const float* ba   = (const float*)d_in[9];
    const float* Vb   = (const float*)d_in[10];
    const float* bb   = (const float*)d_in[11];
    const float* Vc   = (const float*)d_in[12];
    const float* bc   = (const float*)d_in[13];
    const float* rhoW = (const float*)d_in[14];
    const float* rhob = (const float*)d_in[15];
    const float* clsW = (const float*)d_in[16];
    const float* clsb = (const float*)d_in[17];
    float* out = (float*)d_out;

    char* ws = (char*)d_ws;
    int*   order    = (int*)(ws);                                  // 32768 B
    int*   cnt      = (int*)(ws + 32768);                          // 64 B
    int*   offs     = (int*)(ws + 32832);                          // 64 B
    int*   toff     = (int*)(ws + 32896);                          // 64 B
    float* partials = (float*)(ws + 33024);                        // 272*512*4
    float* pooled   = (float*)(ws + 33024 + (size_t)MAX_SLOTS * DH * 4);

    bucket_kernel<<<1, 32, 0, stream>>>(cid, order, cnt, offs, toff);

    dim3 grid(MAX_TILES, KCLUS);
    phi_kernel<<<grid, 256, 0, stream>>>(x, order, cnt, offs, toff,
                                         W1, b1, W2, b2, partials);

    reduce_kernel<<<KCLUS, DH, 0, stream>>>(partials, cnt, toff, pooled);

    tail_kernel<<<1, 512, 0, stream>>>(pooled, fcW, fcb, Va, ba, Vb, bb,
                                       Vc, bc, rhoW, rhob, clsW, clsb, out);

    (void)in_sizes; (void)n_in; (void)out_size; (void)ws_size;
}
